// SparseSelfAttention_30906584662605
// MI455X (gfx1250) — matrix operand order repaired and checked by reading, not compiled, not run on hardware
//
#include <hip/hip_runtime.h>

#define DEV __device__ __forceinline__

typedef unsigned short u16t;
typedef __attribute__((ext_vector_type(16))) __bf16 bf16x16;
typedef __attribute__((ext_vector_type(8)))  float  f32x8;

union FragU { bf16x16 bf; uint4 q[2]; };

static constexpr int Bb   = 4;
static constexpr int S    = 8192;
static constexpr int D    = 1024;
static constexpr int H    = 16;
static constexpr int HD   = 64;   // head dim
static constexpr int BS   = 64;   // attention block size
static constexpr int NB   = S / BS;       // 128
static constexpr int Mtot = Bb * S;       // 32768

// GEMM tiling
static constexpr int TM  = 128;   // rows per WG
static constexpr int TN  = 256;   // cols per WG
static constexpr int TK  = 32;    // K step
static constexpr int LDT = 40;    // padded LDS stride (u16)

// workspace layout (u16 elements)
static constexpr size_t OFF_WT  = 0;                          // 3 * D*D   (W transposed, bf16)
static constexpr size_t OFF_X   = OFF_WT + (size_t)3 * D * D; // Mtot*D    (x, bf16)
static constexpr size_t OFF_QKV = OFF_X + (size_t)Mtot * D;   // 3 * Mtot*D (Q,K,V bf16)

DEV u16t f2bf(float x) {
  union { float f; unsigned int u; } c; c.f = x;
  unsigned int r = c.u + 0x7FFFu + ((c.u >> 16) & 1u);  // round-to-nearest-even
  return (u16t)(r >> 16);
}

DEV f32x8 wmma_bf16(const FragU& a, const FragU& b, f32x8 c) {
  return __builtin_amdgcn_wmma_f32_16x16x32_bf16(false, a.bf, false, b.bf,
                                                 (short)0, c, false, false);
}

// A fragment: 16 rows x 32 K, LDS row-major (stride ld u16).
// lane l: row = l%16, K = {lk..lk+7, lk+16..lk+23}, lk = (l/16)*8.
DEV void load_a(FragU& fr, const u16t* base, int ld, int lane) {
  const u16t* p = base + (lane & 15) * ld + ((lane >> 4) << 3);
  fr.q[0] = *(const uint4*)p;
  fr.q[1] = *(const uint4*)(p + 16);
}

// B fragment: 32 K x 16 cols, LDS holds B TRANSPOSED: [col][K] row-major.
// lane l: col = l%16, K = {lk..lk+7, lk+16..lk+23}, lk = (l/16)*8.
DEV void load_b(FragU& fr, const u16t* base, int ld, int lane) {
  const u16t* p = base + (lane & 15) * ld + ((lane >> 4) << 3);
  fr.q[0] = *(const uint4*)p;
  fr.q[1] = *(const uint4*)(p + 16);
}

// CDNA5 async DMA: global -> LDS, 16 bytes per lane, tracked by ASYNCcnt.
// ldsaddr: LDS byte offset (low 32 bits of generic pointer to __shared__).
DEV void async_ld_b128(unsigned ldsaddr, const void* gaddr) {
  asm volatile("global_load_async_to_lds_b128 %0, %1, off"
               :: "v"(ldsaddr), "v"(gaddr) : "memory");
}
DEV void wait_async(int n) {
  if (n == 0) asm volatile("s_wait_asynccnt 0x0" ::: "memory");
  else        asm volatile("s_wait_asynccnt 0x6" ::: "memory");
}

// ---------------- kernel 0a: x fp32 -> bf16 ----------------
__global__ __launch_bounds__(256) void conv_x_kernel(const float* __restrict__ x,
                                                     u16t* __restrict__ xb) {
  size_t i = ((size_t)blockIdx.x * 256 + threadIdx.x) * 4;
  float4 v = *(const float4*)(x + i);
  union { u16t s[4]; uint2 d; } u;
  u.s[0] = f2bf(v.x); u.s[1] = f2bf(v.y); u.s[2] = f2bf(v.z); u.s[3] = f2bf(v.w);
  *(uint2*)(xb + i) = u.d;
}

// ---------------- kernel 0b: W fp32 -> transposed bf16 ----------------
__global__ __launch_bounds__(256) void conv_w_kernel(const float* __restrict__ Wq,
                                                     const float* __restrict__ Wk,
                                                     const float* __restrict__ Wv,
                                                     u16t* __restrict__ wt) {
  int z = blockIdx.y;
  const float* W = (z == 0) ? Wq : (z == 1) ? Wk : Wv;
  u16t* o = wt + (size_t)z * D * D;
  int e = blockIdx.x * 256 + threadIdx.x;    // coalesced read
  int k = e >> 10;
  int n = e & (D - 1);
  o[(size_t)n * D + k] = f2bf(W[e]);         // scattered write, L2-resident (6 MB)
}

// ---------------- kernel 1: QKV projection GEMM ----------------
// C_bf16[z][Mtot,D] = x_bf16[Mtot,D] @ W_bf16[z] (stored transposed) + bias[z]
// 8 waves in 2x4; each wave owns a 64x64 tile: 16 WMMA per 8 frag loads.
// Staging: double-buffered async global->LDS DMA (6 x b128 per thread per step).
__global__ __launch_bounds__(256) void qkv_gemm(const u16t* __restrict__ A,
                                                const u16t* __restrict__ Wt,
                                                const float* __restrict__ bq,
                                                const float* __restrict__ bk,
                                                const float* __restrict__ bv,
                                                u16t* __restrict__ C) {
  const int n0 = blockIdx.x * TN;
  const int m0 = blockIdx.y * TM;
  const int z  = blockIdx.z;
  const u16t* W = Wt + (size_t)z * D * D;
  const float* bias = (z == 0) ? bq : (z == 1) ? bk : bv;
  u16t* Co = C + (size_t)z * Mtot * D;

  __shared__ __align__(16) u16t As[2][TM * LDT];  // 2 x 10 KB
  __shared__ __align__(16) u16t Bs[2][TN * LDT];  // 2 x 20 KB  (transposed weights)

  const int tid  = threadIdx.x;
  const int lane = tid & 31;
  const int wid  = tid >> 5;
  const int wr   = wid >> 2;   // 0..1: 64-row group
  const int wc   = wid & 3;    // 0..3: 64-col group

  // staging assignment: thread t owns 8-u16 chunks with row = t/4 (+64*q), kc = (t%4)*8
  const int ar  = tid >> 2;
  const int akc = (tid & 3) << 3;

  const u16t* gA = &A[(size_t)(m0 + ar) * D + akc];
  const u16t* gB = &W[(size_t)(n0 + ar) * D + akc];

  unsigned lA[2], lB[2];
  lA[0] = (unsigned)(uintptr_t)&As[0][ar * LDT + akc];
  lA[1] = (unsigned)(uintptr_t)&As[1][ar * LDT + akc];
  lB[0] = (unsigned)(uintptr_t)&Bs[0][ar * LDT + akc];
  lB[1] = (unsigned)(uintptr_t)&Bs[1][ar * LDT + akc];
  const unsigned lstep = 64u * LDT * 2u;  // bytes between 64-row chunk groups in LDS

  auto issue = [&](int buf, int kofs) {
    const u16t* ga = gA + kofs;
    const u16t* gb = gB + kofs;
    async_ld_b128(lA[buf],           ga);
    async_ld_b128(lA[buf] + lstep,   ga + (size_t)64 * D);
#pragma unroll
    for (int q = 0; q < 4; q++)
      async_ld_b128(lB[buf] + (unsigned)q * lstep, gb + (size_t)q * 64 * D);
  };

  f32x8 acc[4][4] = {};

  issue(0, 0);
  const int NIT = D / TK;  // 32
  for (int it = 0; it < NIT; ++it) {
    if (it + 1 < NIT) { issue((it + 1) & 1, (it + 1) * TK); wait_async(6); }
    else              { wait_async(0); }
    __syncthreads();

    const u16t* as = As[it & 1];
    const u16t* bs = Bs[it & 1];
    FragU af[4], bf_[4];
#pragma unroll
    for (int i = 0; i < 4; i++) load_a(af[i], &as[(wr * 64 + i * 16) * LDT], LDT, lane);
#pragma unroll
    for (int j = 0; j < 4; j++) load_b(bf_[j], &bs[(wc * 64 + j * 16) * LDT], LDT, lane);
#pragma unroll
    for (int i = 0; i < 4; i++)
#pragma unroll
      for (int j = 0; j < 4; j++)
        acc[i][j] = wmma_bf16(af[i], bf_[j], acc[i][j]);

    __syncthreads();  // all frag reads done before this buffer is overwritten
  }

  const int cr = (lane >> 4) * 8;
#pragma unroll
  for (int j = 0; j < 4; j++) {
    int n = n0 + wc * 64 + j * 16 + (lane & 15);
    float bia = bias[n];
#pragma unroll
    for (int i = 0; i < 4; i++) {
      size_t mrow = (size_t)m0 + wr * 64 + i * 16 + cr;
#pragma unroll
      for (int e = 0; e < 8; e++)
        Co[(mrow + e) * D + n] = f2bf(acc[i][j][e] + bia);
    }
  }
}

// ---------------- kernel 2: block-diagonal attention ----------------
// one workgroup (4 waves) per (head, block, batch); 64x64 tiles
__global__ __launch_bounds__(128) void block_attn(const u16t* __restrict__ Qb,
                                                  const u16t* __restrict__ Kb,
                                                  const u16t* __restrict__ Vb,
                                                  float* __restrict__ out) {
  const int h   = blockIdx.x;
  const int blk = blockIdx.y;
  const int b   = blockIdx.z;
  const size_t row0 = (size_t)b * S + (size_t)blk * BS;
  const int col0 = h * HD;

  __shared__ __align__(16) u16t Qs[BS * 72];
  __shared__ __align__(16) u16t Ks[BS * 72];
  __shared__ __align__(16) u16t Vt[HD * 72];  // V transposed: [dd][j]
  __shared__ __align__(16) u16t Ps[BS * 72];  // softmax probs, bf16
  __shared__ float Ss[BS * 68];               // scores fp32

  const int tid  = threadIdx.x;
  const int lane = tid & 31;
  const int w    = tid >> 5;  // wave: rows w*16..w*16+15

  // stage Q,K rows and V transposed
  for (int c = tid; c < 512; c += 128) {
    int i  = c >> 3;
    int d0 = (c & 7) << 3;
    size_t g = (row0 + i) * D + col0 + d0;
    *(uint4*)&Qs[i * 72 + d0] = *(const uint4*)&Qb[g];
    *(uint4*)&Ks[i * 72 + d0] = *(const uint4*)&Kb[g];
    uint4 vv = *(const uint4*)&Vb[g];
    const u16t* pv = (const u16t*)&vv;
#pragma unroll
    for (int e = 0; e < 8; e++) Vt[(d0 + e) * 72 + i] = pv[e];
  }
  __syncthreads();

  // S = Q @ K^T : B-fragment of K^T == row reads of K
  f32x8 sacc[4] = {};
#pragma unroll
  for (int k0 = 0; k0 < HD; k0 += 32) {
    FragU a; load_a(a, &Qs[(w * 16) * 72 + k0], 72, lane);
#pragma unroll
    for (int j = 0; j < 4; j++) {
      FragU bf_; load_b(bf_, &Ks[(j * 16) * 72 + k0], 72, lane);
      sacc[j] = wmma_bf16(a, bf_, sacc[j]);
    }
  }
  {
    const int cr = (lane >> 4) * 8;
#pragma unroll
    for (int j = 0; j < 4; j++) {
      int col = j * 16 + (lane & 15);
#pragma unroll
      for (int e = 0; e < 8; e++)
        Ss[(w * 16 + cr + e) * 68 + col] = sacc[j][e] * 0.125f;  // 1/sqrt(64)
    }
  }
  __syncthreads();

  // per-row softmax (64 rows; threads 0..63 each own a row)
  if (tid < BS) {
    float* r = &Ss[tid * 68];
    float mx = r[0];
    for (int j = 1; j < BS; j++) mx = fmaxf(mx, r[j]);
    float sum = 0.f;
    for (int j = 0; j < BS; j++) { float e = __expf(r[j] - mx); r[j] = e; sum += e; }
    float inv = 1.0f / sum;
    for (int j = 0; j < BS; j++) Ps[tid * 72 + j] = f2bf(r[j] * inv);
  }
  __syncthreads();

  // O = P @ V : B-fragment reads rows of Vt (= columns of V)
  f32x8 oacc[4] = {};
#pragma unroll
  for (int k0 = 0; k0 < BS; k0 += 32) {
    FragU a; load_a(a, &Ps[(w * 16) * 72 + k0], 72, lane);
#pragma unroll
    for (int j = 0; j < 4; j++) {
      FragU bf_; load_b(bf_, &Vt[(j * 16) * 72 + k0], 72, lane);
      oacc[j] = wmma_bf16(a, bf_, oacc[j]);
    }
  }
  {
    const int cr = (lane >> 4) * 8;
#pragma unroll
    for (int j = 0; j < 4; j++) {
      int dd = j * 16 + (lane & 15);
#pragma unroll
      for (int e = 0; e < 8; e++)
        out[(row0 + w * 16 + cr + e) * D + col0 + dd] = oacc[j][e];
    }
  }
}

extern "C" void kernel_launch(void* const* d_in, const int* in_sizes, int n_in,
                              void* d_out, int out_size, void* d_ws, size_t ws_size,
                              hipStream_t stream) {
  (void)in_sizes; (void)n_in; (void)out_size; (void)ws_size;
  const float* x  = (const float*)d_in[0];
  const float* Wq = (const float*)d_in[1];
  const float* bq = (const float*)d_in[2];
  const float* Wk = (const float*)d_in[3];
  const float* bk = (const float*)d_in[4];
  const float* Wv = (const float*)d_in[5];
  const float* bv = (const float*)d_in[6];
  float* out = (float*)d_out;
  u16t* wsu  = (u16t*)d_ws;

  // 0a: x -> bf16
  conv_x_kernel<<<(unsigned)((size_t)Mtot * D / 1024), 256, 0, stream>>>(x, wsu + OFF_X);
  // 0b: W -> transposed bf16
  conv_w_kernel<<<dim3(D * D / 256, 3, 1), 256, 0, stream>>>(Wq, Wk, Wv, wsu + OFF_WT);
  // 1: QKV projection GEMM (WMMA bf16, async double-buffered staging)
  qkv_gemm<<<dim3(D / TN, Mtot / TM, 3), 256, 0, stream>>>(
      wsu + OFF_X, wsu + OFF_WT, bq, bk, bv, wsu + OFF_QKV);
  // 2: block attention (WMMA bf16)
  block_attn<<<dim3(H, NB, Bb), 128, 0, stream>>>(
      wsu + OFF_QKV,
      wsu + OFF_QKV + (size_t)Mtot * D,
      wsu + OFF_QKV + (size_t)2 * Mtot * D,
      out);
}
